// GauntTensorProductAllParitiesS2Grid_11209864642659
// MI455X (gfx1250) — matrix operand order repaired and checked by reading, not compiled, not run on hardware
//
#include <hip/hip_runtime.h>
#include <hip/hip_bf16.h>

typedef _Float16 v16h __attribute__((ext_vector_type(16)));
typedef _Float16 v8h  __attribute__((ext_vector_type(8)));
typedef float    v8f  __attribute__((ext_vector_type(8)));

#define N_TOT   2048
#define MUL     64
#define CCH     64
#define GRID_BA 780      // 20 * 39
#define GRID_PAD 800     // padded to 25 chunks of 32

__device__ __forceinline__ v8f wmma_f16(v16h a, v16h b, v8f c) {
    return __builtin_amdgcn_wmma_f32_16x16x32_f16(false, a, false, b, (short)0, c, false, false);
}

// ---------------------------------------------------------------------------
// Prep kernel: pack all static operands into wave32 WMMA B-operand layout
// (col = lane&15, k = (lane>>4)*16 + e; YinOp: k = 8*(lane>>4) + (e&7) + 16*(e>>3)), fp32 -> f16, scale factors folded in.
// ---------------------------------------------------------------------------
__global__ void prep_kernel(const float* __restrict__ W1, const float* __restrict__ W2,
                            const float* __restrict__ Wout,
                            const float* __restrict__ Yin, const float* __restrict__ Yout,
                            const float* __restrict__ qw,
                            _Float16* __restrict__ w1op, _Float16* __restrict__ w2op,
                            _Float16* __restrict__ yinop, _Float16* __restrict__ youtop,
                            _Float16* __restrict__ woutop) {
    int tid = blockIdx.x * blockDim.x + threadIdx.x;
    int nth = gridDim.x * blockDim.x;

    // W1Op / W2Op: [b][l][ct][ks][lane][e]  (3*3*4*2*32*16 = 36864)
    for (int i = tid; i < 36864; i += nth) {
        int e = i & 15, lane = (i >> 4) & 31, ks = (i >> 9) & 1, ct = (i >> 10) & 3;
        int l = (i >> 12) % 3, b = (i >> 12) / 3;
        int cc = ct * 16 + (lane & 15);
        int m  = ks * 32 + (lane >> 4) * 16 + e;
        w1op[i] = (_Float16)(W1[((b * 3 + l) * MUL + m) * CCH + cc] * 0.125f);
        w2op[i] = (_Float16)(W2[((b * 3 + l) * MUL + m) * CCH + cc] * 0.125f);
    }
    // YinOp: [gt(50)][lane][e]   cols = grid g, k = lm (only 0..8 valid)
    for (int i = tid; i < 50 * 512; i += nth) {
        int e = i & 15, lane = (i >> 4) & 31, gt = i >> 9;
        int k = (lane >> 4) * 8 + (e & 7) + (e >= 8 ? 16 : 0);
        int g = gt * 16 + (lane & 15);
        float v = (k < 9 && g < GRID_BA) ? Yin[k * GRID_BA + g] : 0.0f;
        yinop[i] = (_Float16)v;
    }
    // YoutOp: [gc(25)][ct(2)][lane][e]  cols = l' (25, pad 32), k = g within chunk, qw folded
    for (int i = tid; i < 25 * 2 * 512; i += nth) {
        int e = i & 15, lane = (i >> 4) & 31, ct = (i >> 9) & 1, gc = i >> 10;
        int lp = ct * 16 + (lane & 15);
        int k  = (lane >> 4) * 16 + e;
        int g  = gc * 32 + k;
        float v = (lp < 25 && g < GRID_BA) ? Yout[lp * GRID_BA + g] * qw[g / 39] : 0.0f;
        youtop[i] = (_Float16)v;
    }
    // WoutOp: [b][l(5)][dt(4)][ks(2)][lane][e]  cols = d, k = c, 1/8 folded
    for (int i = tid; i < 3 * 5 * 4 * 2 * 512; i += nth) {
        int e = i & 15, lane = (i >> 4) & 31, ks = (i >> 9) & 1, dt = (i >> 10) & 3;
        int l = (i >> 12) % 5, b = (i >> 12) / 5;
        int d = dt * 16 + (lane & 15);
        int c = ks * 32 + (lane >> 4) * 16 + e;
        woutop[i] = (_Float16)(Wout[((b * 5 + l) * CCH + c) * CCH + d] * 0.125f);
    }
}

// ---------------------------------------------------------------------------
// Stage A: projection c{1,2}[n][b][c][16] = (x · W) / 8, per (branch, lm).
// GEMM rows=n(16), cols=c(16), K=m(64, 2 WMMA k-steps). 8 waves/block:
// wave = (tensorSel, ctile).
// ---------------------------------------------------------------------------
__global__ void stageA_kernel(const float* __restrict__ x1, const float* __restrict__ x2,
                              const _Float16* __restrict__ w1op, const _Float16* __restrict__ w2op,
                              _Float16* __restrict__ c1, _Float16* __restrict__ c2) {
    int lane = threadIdx.x & 31, wv = threadIdx.x >> 5;
    int lr = lane & 15, hi = lane >> 4;
    int tsel = wv >> 2, ct = wv & 3;

    int blk = blockIdx.x;
    int ntile = blk % 128; blk /= 128;
    int lm = blk % 9;
    int b  = blk / 9;
    int l  = (lm == 0) ? 0 : (lm < 4 ? 1 : 2);
    int p  = tsel ? ((b == 1) ? -1 : 1) : ((b == 2) ? -1 : 1);   // p2 : p1 per branch
    int pidx = (p == 1) ? (l & 1) : ((l & 1) ^ 1);

    const float*    xs  = tsel ? x2 : x1;
    const _Float16* wop = tsel ? w2op : w1op;
    _Float16*       cd  = tsel ? c2 : c1;

    int n = ntile * 16 + lr;   // A row
    v8f acc;
    #pragma unroll
    for (int i = 0; i < 8; i++) acc[i] = 0.0f;

    #pragma unroll
    for (int ks = 0; ks < 2; ks++) {
        v16h A;
        #pragma unroll
        for (int e = 0; e < 16; e++) {
            int k = hi * 16 + e;
            int m = ks * 32 + k;
            A[e] = (_Float16)xs[((n * MUL + m) * 2 + pidx) * 9 + lm];
        }
        v16h B = *(const v16h*)(wop + ((((b * 3 + l) * 4 + ct) * 2 + ks) * 32 + lane) * 16);
        acc = wmma_f16(A, B, acc);
    }
    int c = ct * 16 + lr;
    #pragma unroll
    for (int v = 0; v < 8; v++) {
        int nn = ntile * 16 + v + 8 * hi;
        cd[(((long)nn * 3 + b) * CCH + c) * 16 + lm] = (_Float16)acc[v];
    }
}

// ---------------------------------------------------------------------------
// Fused stages B+C+D: per wave, 16 (n,c) rows; stream grid in chunks of 32:
//   S1 = c1·Yin, S2 = c2·Yin (4 WMMA), P = S1*S2 (VALU), transpose C/D->A
//   layout through per-wave LDS, then co += P·YoutOp (2 WMMA).
// ---------------------------------------------------------------------------
__global__ void fusedBCD_kernel(const _Float16* __restrict__ c1, const _Float16* __restrict__ c2,
                                const _Float16* __restrict__ yinop, const _Float16* __restrict__ youtop,
                                _Float16* __restrict__ co) {
    __shared__ _Float16 lds[8][16][32];
    int lane = threadIdx.x & 31, wv = threadIdx.x >> 5;
    int lr = lane & 15, hi = lane >> 4;

    int wid = blockIdx.x * 8 + wv;           // 24576 wave tasks
    int b = wid / 8192;
    int r = wid % 8192;
    int n = r >> 2, cblk = r & 3;
    int c = cblk * 16 + lr;
    long rowbase = ((long)(n * 3 + b) * CCH + c) * 16;

    // Persistent A operands (K = lm padded to 32; only lm 0..8 nonzero)
    v16h A1, A2;
    {
        v8h a1 = *(const v8h*)(c1 + rowbase + hi * 8);
        v8h a2 = *(const v8h*)(c2 + rowbase + hi * 8);
        #pragma unroll
        for (int e = 0; e < 16; e++) { A1[e] = (_Float16)0.0f; A2[e] = (_Float16)0.0f; }
        #pragma unroll
        for (int e = 0; e < 8; e++) {
            bool valid = (hi == 0) || (e == 0);   // k = 8*hi + e < 9
            A1[e] = valid ? a1[e] : (_Float16)0.0f;
            A2[e] = valid ? a2[e] : (_Float16)0.0f;
        }
    }

    v8f acc0, acc1;
    #pragma unroll
    for (int i = 0; i < 8; i++) { acc0[i] = 0.0f; acc1[i] = 0.0f; }

    for (int gc = 0; gc < 25; gc++) {
        v16h by0 = *(const v16h*)(yinop + (((gc * 2 + 0) * 32) + lane) * 16);
        v16h by1 = *(const v16h*)(yinop + (((gc * 2 + 1) * 32) + lane) * 16);
        v8f z;
        #pragma unroll
        for (int i = 0; i < 8; i++) z[i] = 0.0f;
        v8f s1a = wmma_f16(A1, by0, z);
        v8f s1b = wmma_f16(A1, by1, z);
        v8f s2a = wmma_f16(A2, by0, z);
        v8f s2b = wmma_f16(A2, by1, z);

        // pointwise product, stash in per-wave LDS tile [row 16][k 32]
        #pragma unroll
        for (int v = 0; v < 8; v++) {
            lds[wv][v + 8 * hi][lr]      = (_Float16)(s1a[v] * s2a[v]);
            lds[wv][v + 8 * hi][16 + lr] = (_Float16)(s1b[v] * s2b[v]);
        }
        // read back in A-operand layout: k = 16*hi+e
        v8h lo  = *(const v8h*)&lds[wv][lr][16 * hi];
        v8h hig = *(const v8h*)&lds[wv][lr][16 * hi + 8];
        v16h Ap;
        #pragma unroll
        for (int e = 0; e < 8; e++) { Ap[e] = lo[e]; Ap[8 + e] = hig[e]; }

        v16h byo0 = *(const v16h*)(youtop + (((gc * 2 + 0) * 32) + lane) * 16);
        v16h byo1 = *(const v16h*)(youtop + (((gc * 2 + 1) * 32) + lane) * 16);
        acc0 = wmma_f16(Ap, byo0, acc0);
        acc1 = wmma_f16(Ap, byo1, acc1);
    }

    long obase = (long)(n * 3 + b) * CCH;
    #pragma unroll
    for (int v = 0; v < 8; v++) {
        int cc = cblk * 16 + v + 8 * hi;
        co[(obase + cc) * 32 + lr]      = (_Float16)acc0[v];
        co[(obase + cc) * 32 + 16 + lr] = (_Float16)acc1[v];
    }
}

// ---------------------------------------------------------------------------
// Stage E: out[n][b][d][l'] = sum_c co[n][b][c][l'] * Wout[b][l(l')][c][d] / 8
// Per (b, l', ntile): rows=n(16), cols=d (4 tiles = 4 waves), K=c(64).
// ---------------------------------------------------------------------------
__global__ void stageE_kernel(const _Float16* __restrict__ co,
                              const _Float16* __restrict__ woutop,
                              float* __restrict__ out) {
    int lane = threadIdx.x & 31, wv = threadIdx.x >> 5;   // block = 128, dt = wv
    int lr = lane & 15, hi = lane >> 4;

    int blk = blockIdx.x;
    int ntile = blk % 128; blk /= 128;
    int l = blk % 25;
    int b = blk / 25;
    int lq = (l == 0) ? 0 : (l < 4) ? 1 : (l < 9) ? 2 : (l < 16) ? 3 : 4;
    int dt = wv;

    v8f acc;
    #pragma unroll
    for (int i = 0; i < 8; i++) acc[i] = 0.0f;

    #pragma unroll
    for (int ks = 0; ks < 2; ks++) {
        v16h A;
        int nn = ntile * 16 + lr;
        #pragma unroll
        for (int e = 0; e < 16; e++) {
            int k  = hi * 16 + e;
            int cc = ks * 32 + k;
            A[e] = co[((long)(nn * 3 + b) * CCH + cc) * 32 + l];
        }
        v16h B = *(const v16h*)(woutop + ((((b * 5 + lq) * 4 + dt) * 2 + ks) * 32 + lane) * 16);
        acc = wmma_f16(A, B, acc);
    }
    #pragma unroll
    for (int v = 0; v < 8; v++) {
        int nn = ntile * 16 + v + 8 * hi;
        int d  = dt * 16 + lr;
        out[((long)(nn * 3 + b) * CCH + d) * 25 + l] = acc[v];
    }
}

// ---------------------------------------------------------------------------
extern "C" void kernel_launch(void* const* d_in, const int* in_sizes, int n_in,
                              void* d_out, int out_size, void* d_ws, size_t ws_size,
                              hipStream_t stream) {
    const float* x1   = (const float*)d_in[0];
    const float* x2   = (const float*)d_in[1];
    const float* W1   = (const float*)d_in[2];
    const float* W2   = (const float*)d_in[3];
    const float* Wout = (const float*)d_in[4];
    const float* Yin  = (const float*)d_in[5];
    const float* Yout = (const float*)d_in[6];
    const float* qw   = (const float*)d_in[7];
    float* out = (float*)d_out;

    char* ws = (char*)d_ws;
    const size_t sz_c  = (size_t)N_TOT * 3 * CCH * 16 * sizeof(_Float16);  // 12,582,912
    const size_t sz_co = (size_t)N_TOT * 3 * CCH * 32 * sizeof(_Float16);  // 25,165,824
    _Float16* c1     = (_Float16*)(ws);
    _Float16* c2     = (_Float16*)(ws + sz_c);
    _Float16* cobuf  = (_Float16*)(ws + 2 * sz_c);
    _Float16* w1op   = (_Float16*)(ws + 2 * sz_c + sz_co);
    _Float16* w2op   = (_Float16*)(ws + 2 * sz_c + sz_co + 73728);
    _Float16* yinop  = (_Float16*)(ws + 2 * sz_c + sz_co + 2 * 73728);
    _Float16* youtop = (_Float16*)(ws + 2 * sz_c + sz_co + 2 * 73728 + 51200);
    _Float16* woutop = (_Float16*)(ws + 2 * sz_c + sz_co + 2 * 73728 + 2 * 51200);

    prep_kernel<<<96, 256, 0, stream>>>(W1, W2, Wout, Yin, Yout, qw,
                                        w1op, w2op, yinop, youtop, woutop);
    stageA_kernel<<<3 * 9 * 128, 256, 0, stream>>>(x1, x2, w1op, w2op, c1, c2);
    fusedBCD_kernel<<<3072, 256, 0, stream>>>(c1, c2, yinop, youtop, cobuf);
    stageE_kernel<<<3 * 25 * 128, 128, 0, stream>>>(cobuf, woutop, out);
}
